// MSNet_28621662060965
// MI455X (gfx1250) — compile-verified, matrix operand order checked by reading, not run on hardware
//
#include <hip/hip_runtime.h>
#include <hip/hip_bf16.h>

// ---------------- model constants ----------------
#define B_    8
#define T_    1000
#define E_    32
#define DF_   96
#define FREQ_ 481
#define C_    16
#define DH_   64
#define HS_   16

typedef _Float16 half_t;
typedef __attribute__((ext_vector_type(16))) _Float16 v16h;
typedef __attribute__((ext_vector_type(8)))  float    v8f;

__device__ inline float sigm(float x) { return 1.f / (1.f + expf(-x)); }
__device__ inline float gelu_f(float x) {
  float x3 = x * x * x;
  return 0.5f * x * (1.f + tanhf(0.7978845608028654f * (x + 0.044715f * x3)));
}
__device__ inline v8f zero8() { v8f z = {0.f,0.f,0.f,0.f,0.f,0.f,0.f,0.f}; return z; }

// ---------------- CDNA5 WMMA fragment layout helpers -----------------------
// 16x32 f16 A-fragment (ISA 7.12.2): lane l<16 holds row M=l, K in {0..7,16..23};
// lane l>=16 holds M=l-16, K in {8..15,24..31}; 16 halves per lane.
// Inverse map of element (row r, col k) -> flat index into [32 lanes][16 halves]:
__device__ inline int frag_idx(int r, int k) {
  int lane = (r & 15) | ((k & 8) ? 16 : 0);
  int h    = (k & 7) | ((k & 16) ? 8 : 0);
  return lane * 16 + h;
}

// Both operands stored in fragment layout: each lane does 2x b128 loads.
__device__ inline v8f wmma_frag(const half_t* __restrict__ A,
                                const half_t* __restrict__ Bf, v8f acc) {
  const int lane = threadIdx.x & 31;
  v16h a = *(const v16h*)(A + lane * 16);
  v16h b = *(const v16h*)(Bf + lane * 16);
  return __builtin_amdgcn_wmma_f32_16x16x32_f16(false, a, false, b, (short)0, acc,
                                                false, false);
}

// D fragment (f32 16x16): VGPR r holds M=r (lanes 0-15) / M=r+8 (lanes 16-31), N=lane&15.
template <typename FS>
__device__ inline void tile_store(const v8f& d, const FS& st) {
  const int lane = threadIdx.x & 31;
  const int n  = lane & 15;
  const int mb = (lane & 16) ? 8 : 0;
#pragma unroll
  for (int r = 0; r < 8; ++r) st(mb + r, n, d[r]);
}

// ---------------- async LDS fill (ASYNCcnt path, ISA 15.18.3) --------------
// nbytes must be a multiple of 32 lanes * 16B = 512B (true for F=32/96 tiles).
__device__ inline void async_copy_frame(const float* __restrict__ src,
                                        float* dstLds, int nbytes) {
  const int lane = threadIdx.x & 31;
  // generic LDS address: low 32 bits are the LDS byte offset
  uint32_t base = (uint32_t)(uintptr_t)dstLds;
  for (int off = lane * 16; off < nbytes; off += 32 * 16) {
    uint32_t laddr = base + (uint32_t)off;
    uint32_t voff  = (uint32_t)off;
    asm volatile("global_load_async_to_lds_b128 %0, %1, %2 offset:0"
                 :: "v"(laddr), "v"(voff), "s"(src) : "memory");
  }
  asm volatile("s_wait_asynccnt 0" ::: "memory");
}

// ---------------- weight pre-pack into B-fragment layout -------------------
// src: f32 [K,N] row-major; dst: tiles (kc,nt) of 512 halves, kc-major.
__global__ void k_pack_b(const float* __restrict__ src, half_t* __restrict__ dst,
                         int K, int N) {
  const int Nt = N >> 4;
  const int kc = blockIdx.x / Nt, nt = blockIdx.x % Nt;
  const int tid = threadIdx.x;                 // 512 threads per tile
  const int lane = tid >> 4, h = tid & 15;
  const int k = kc * 32 + ((h & 7) | ((lane & 16) ? 8 : 0) | ((h & 8) ? 16 : 0));
  const int n = nt * 16 + (lane & 15);
  dst[(size_t)blockIdx.x * 512 + tid] = (k < K) ? (half_t)src[(size_t)k * N + n]
                                                : (half_t)0.f;
}

// ---------------- ERB input: LN(E) -> conv1x1(1->16) -> gelu ---------------
__global__ void k_erb_in(const float* __restrict__ feat, const float* __restrict__ g,
                         const float* __restrict__ be, const float* __restrict__ Wci,
                         const float* __restrict__ bci, float* __restrict__ X) {
  const int bt = blockIdx.x, lane = threadIdx.x;      // lane == erb bin e
  float v = feat[(size_t)bt * E_ + lane];
  float s = v, s2 = v * v;
  for (int o = 16; o > 0; o >>= 1) { s += __shfl_xor(s, o, 32); s2 += __shfl_xor(s2, o, 32); }
  float mu  = s / E_;
  float var = s2 / E_ - mu * mu;
  float ln  = (v - mu) * rsqrtf(var + 1e-5f) * g[lane] + be[lane];
  float* xp = X + ((size_t)bt * E_ + lane) * C_;
#pragma unroll
  for (int c = 0; c < C_; ++c) xp[c] = gelu_f(ln * Wci[c] + bci[c]);
}

// ---------------- pointwise ConvGRU over time (state per (b,f) row) --------
__global__ void k_gru(float* __restrict__ X, const float* __restrict__ Wi,
                      const float* __restrict__ Wh, const float* __restrict__ mask,
                      int F) {
  __shared__ float sWi[C_ * 3 * C_], sWh[C_ * 3 * C_];
  const int tid = threadIdx.x;
  for (int i = tid; i < C_ * 3 * C_; i += blockDim.x) { sWi[i] = Wi[i]; sWh[i] = Wh[i]; }
  __syncthreads();
  const int row = blockIdx.x * blockDim.x + tid;
  if (row >= B_ * F) return;
  const int b = row / F, f = row % F;
  float h[C_];
#pragma unroll
  for (int c = 0; c < C_; ++c) h[c] = 0.f;
  for (int t = 0; t < T_; ++t) {
    float* xp = X + (((size_t)b * T_ + t) * F + f) * C_;
    float xv[C_];
#pragma unroll
    for (int c = 0; c < C_; ++c) xv[c] = xp[c];
    if (t + 1 < T_) __builtin_prefetch(xp + (size_t)F * C_, 0, 3);  // global_prefetch_b8
    float hn[C_];
#pragma unroll
    for (int j = 0; j < C_; ++j) {
      float a0 = 0, a1 = 0, a2 = 0, b0 = 0, b1 = 0, b2 = 0;
#pragma unroll
      for (int c = 0; c < C_; ++c) {
        float xc = xv[c], hc = h[c];
        a0 += xc * sWi[c * 48 + j];       b0 += hc * sWh[c * 48 + j];
        a1 += xc * sWi[c * 48 + 16 + j];  b1 += hc * sWh[c * 48 + 16 + j];
        a2 += xc * sWi[c * 48 + 32 + j];  b2 += hc * sWh[c * 48 + 32 + j];
      }
      float r = sigm(a0 + b0), z = sigm(a1 + b1);
      float n = tanhf(a2 + r * b2);
      hn[j] = (1.f - z) * n + z * h[j];
    }
    const float mk = mask ? mask[b * T_ + t] : 1.f;
#pragma unroll
    for (int c = 0; c < C_; ++c) {
      h[c]  = (mk > 0.5f) ? hn[c] : h[c];
      xp[c] = xv[c] + h[c];                 // x = x + conv_gru(x)
    }
  }
}

// ---------------- long/short attention (wave per frame, fragment LDS) ------
__global__ void k_attn(float* __restrict__ X, int F,
                       const float* __restrict__ g, const float* __restrict__ be,
                       const half_t* __restrict__ WqF, const half_t* __restrict__ WkF,
                       const half_t* __restrict__ WvF, const half_t* __restrict__ WoF,
                       const float* __restrict__ Wsh) {
  extern __shared__ char smem[];
  const int nMt = F >> 4, nKc = F >> 5;
  float*  Xf    = (float*)smem;                  // F*16 f32 (residual base)
  float*  stats = Xf + F * C_;                   // 32
  float*  st    = stats + 32;                    // 16*F f32 (score row block)
  half_t* lnA   = (half_t*)(st + 16 * F);        // nMt tiles       (A: LN(x), K=16 pad)
  half_t* qA    = lnA + nMt * 512;               // nMt*2 tiles     (A: q, K=DH)
  half_t* kB    = qA + nMt * 2 * 512;            // nMt*2 tiles     (B: k^T)
  half_t* vB    = kB + nMt * 2 * 512;            // nKc*4 tiles     (B: v)
  half_t* P     = vB + nKc * 4 * 512;            // nMt*nKc tiles   (A: softmax probs)
  half_t* ohA   = P + nMt * nKc * 512;           // nMt*2 tiles     (A: long+short)
  const int lane = threadIdx.x;
  float* xg = X + (size_t)blockIdx.x * F * C_;

  async_copy_frame(xg, Xf, F * C_ * 4);
  for (int i = lane; i < nMt * 512; i += 32) lnA[i] = (half_t)0.f;  // zero K-pad
  __syncthreads();
  if (lane < C_) {                      // LN over frequency, per channel
    float s = 0, s2 = 0;
    for (int f = 0; f < F; ++f) { float v = Xf[f * C_ + lane]; s += v; s2 += v * v; }
    float mu = s / F;
    stats[lane] = mu;
    stats[16 + lane] = rsqrtf(s2 / F - mu * mu + 1e-5f);
  }
  __syncthreads();
  for (int i = lane; i < F * C_; i += 32) {
    int f = i >> 4, c = i & 15;
    float v = (Xf[i] - stats[c]) * stats[16 + c] * g[f] + be[f];
    lnA[(f >> 4) * 512 + frag_idx(f, c)] = (half_t)v;
  }
  __syncthreads();

  // ---- QKV: [F,16] x [16,64] (K padded to 32; B = prepacked weight frags) --
  for (int mt = 0; mt < nMt; ++mt)
    for (int nt = 0; nt < 4; ++nt) {
      const half_t* A = lnA + mt * 512;
      v8f aq = wmma_frag(A, WqF + nt * 512, zero8());
      v8f ak = wmma_frag(A, WkF + nt * 512, zero8());
      v8f av = wmma_frag(A, WvF + nt * 512, zero8());
      tile_store(aq, [&](int m, int n, float v) {
        int f = mt * 16 + m, d = nt * 16 + n;
        qA[((f >> 4) * 2 + (d >> 5)) * 512 + frag_idx(f, d & 31)] = (half_t)v; });
      tile_store(ak, [&](int m, int n, float v) {
        int f = mt * 16 + m, d = nt * 16 + n;          // B-frag: n=f, k=d
        kB[((f >> 4) * 2 + (d >> 5)) * 512 + frag_idx(f, d & 31)] = (half_t)v; });
      tile_store(av, [&](int m, int n, float v) {
        int f = mt * 16 + m, d = nt * 16 + n;          // B-frag: n=d, k=f
        vB[((f >> 5) * 4 + (d >> 4)) * 512 + frag_idx(d, f & 31)] = (half_t)v; });
    }
  __syncthreads();

  // ---- scores q k^T / 8 + row softmax, 16 rows at a time -------------------
  for (int mt = 0; mt < nMt; ++mt) {
    for (int nt = 0; nt < nMt; ++nt) {
      v8f acc = zero8();
      for (int kc = 0; kc < 2; ++kc)
        acc = wmma_frag(qA + (mt * 2 + kc) * 512, kB + (nt * 2 + kc) * 512, acc);
      tile_store(acc, [&](int m, int n, float v) { st[m * F + nt * 16 + n] = v * 0.125f; });
    }
    __syncthreads();
    if (lane < 16) {
      float mx = -3.4e38f;
      for (int gg = 0; gg < F; ++gg) mx = fmaxf(mx, st[lane * F + gg]);
      float ss = 0.f;
      for (int gg = 0; gg < F; ++gg) { float e = expf(st[lane * F + gg] - mx); st[lane * F + gg] = e; ss += e; }
      float inv = 1.f / ss;
      for (int gg = 0; gg < F; ++gg)
        P[(mt * nKc + (gg >> 5)) * 512 + frag_idx(lane, gg & 31)] =
            (half_t)(st[lane * F + gg] * inv);
    }
    __syncthreads();
  }

  // ---- long = P @ V, fused with short depthwise conv on V ------------------
  for (int mt = 0; mt < nMt; ++mt)
    for (int nt = 0; nt < 4; ++nt) {
      v8f acc = zero8();
      for (int kc = 0; kc < nKc; ++kc)
        acc = wmma_frag(P + (mt * nKc + kc) * 512, vB + (kc * 4 + nt) * 512, acc);
      tile_store(acc, [&](int m, int n, float v) {
        int f = mt * 16 + m, d = nt * 16 + n;
        float sh = 0.f;
#pragma unroll
        for (int j = 0; j < 9; ++j) {
          int ff = f + j - 4;
          if (ff >= 0 && ff < F)
            sh += (float)vB[((ff >> 5) * 4 + (d >> 4)) * 512 + frag_idx(d, ff & 31)] *
                  Wsh[d * 9 + j];
        }
        ohA[((f >> 4) * 2 + (d >> 5)) * 512 + frag_idx(f, d & 31)] = (half_t)(v + sh);
      });
    }
  __syncthreads();

  // ---- output projection [F,64] x [64,16] + residual -----------------------
  for (int mt = 0; mt < nMt; ++mt) {
    v8f acc = zero8();
    for (int kc = 0; kc < 2; ++kc)
      acc = wmma_frag(ohA + (mt * 2 + kc) * 512, WoF + kc * 512, acc);
    tile_store(acc, [&](int m, int n, float v) {
      int i = (mt * 16 + m) * C_ + n;
      xg[i] = Xf[i] + v;
    });
  }
}

// ---------------- MLP block (wave per frame, fragment LDS) -----------------
__global__ void k_mlp(float* __restrict__ X, int F,
                      const float* __restrict__ g, const float* __restrict__ be,
                      const half_t* __restrict__ Wm1F, const float* __restrict__ bm1,
                      const half_t* __restrict__ Wm2F, const float* __restrict__ bm2) {
  extern __shared__ char smem[];
  const int nMt = F >> 4;
  float*  Xf    = (float*)smem;                // F*16
  float*  stats = Xf + F * C_;                 // 32
  half_t* lnA   = (half_t*)(stats + 32);       // nMt tiles
  half_t* hA    = lnA + nMt * 512;             // nMt tiles
  const int lane = threadIdx.x;
  float* xg = X + (size_t)blockIdx.x * F * C_;
  async_copy_frame(xg, Xf, F * C_ * 4);
  for (int i = lane; i < 2 * nMt * 512; i += 32) lnA[i] = (half_t)0.f;  // zero pads
  __syncthreads();
  if (lane < C_) {
    float s = 0, s2 = 0;
    for (int f = 0; f < F; ++f) { float v = Xf[f * C_ + lane]; s += v; s2 += v * v; }
    float mu = s / F;
    stats[lane] = mu;
    stats[16 + lane] = rsqrtf(s2 / F - mu * mu + 1e-5f);
  }
  __syncthreads();
  for (int i = lane; i < F * C_; i += 32) {
    int f = i >> 4, c = i & 15;
    float v = (Xf[i] - stats[c]) * stats[16 + c] * g[f] + be[f];
    lnA[(f >> 4) * 512 + frag_idx(f, c)] = (half_t)v;
  }
  __syncthreads();
  for (int mt = 0; mt < nMt; ++mt) {
    v8f acc = wmma_frag(lnA + mt * 512, Wm1F, zero8());
    tile_store(acc, [&](int m, int n, float v) {
      hA[mt * 512 + frag_idx(m, n)] = (half_t)gelu_f(v + bm1[n]); });
  }
  __syncthreads();
  for (int mt = 0; mt < nMt; ++mt) {
    v8f acc = wmma_frag(hA + mt * 512, Wm2F, zero8());
    tile_store(acc, [&](int m, int n, float v) {
      int i = (mt * 16 + m) * C_ + n;
      xg[i] = Xf[i] + v + bm2[n];
    });
  }
}

// ---------------- SNR feature projection: [T*B,512] x [512,48] -------------
__global__ void k_snr_xw(const float* __restrict__ X, const half_t* __restrict__ WiF,
                         const float* __restrict__ bi, float* __restrict__ xw) {
  const int mt = blockIdx.x, nt = blockIdx.y;
  const int lane = threadIdx.x;
  const int m = lane & 15, kbase = (lane & 16) ? 8 : 0;
  v8f acc = zero8();
  for (int kc = 0; kc < 16; ++kc) {
    union { v16h v; half_t e[16]; } A;
#pragma unroll
    for (int h = 0; h < 16; ++h) {
      const int vg = h >> 1, p = h & 1;
      const int k = (vg < 4) ? (kbase + 2 * vg + p) : (16 + kbase + 2 * (vg - 4) + p);
      int row = mt * 16 + m;            // row = t*B + b
      int t = row >> 3, b = row & 7;
      int feat = kc * 32 + k;           // feat = c*E + e
      int c = feat >> 5, e = feat & 31;
      A.e[h] = (half_t)X[(((size_t)b * T_ + t) * E_ + e) * C_ + c];
    }
    v16h Bv = *(const v16h*)(WiF + ((size_t)(kc * 3 + nt)) * 512 + lane * 16);
    acc = __builtin_amdgcn_wmma_f32_16x16x32_f16(false, A.v, false, Bv, (short)0, acc,
                                                 false, false);
  }
  tile_store(acc, [&](int mm, int n, float v) {
    int row = mt * 16 + mm, col = nt * 16 + n;
    xw[(size_t)row * 48 + col] = v + bi[col];
  });
}

// ---------------- SNR GRU (HS=16) + lsnr head ------------------------------
__global__ void k_snr_gru(const float* __restrict__ xw, const float* __restrict__ Wh,
                          const float* __restrict__ bh, const float* __restrict__ fcW,
                          const float* __restrict__ fcb, float* __restrict__ lsnr_ws,
                          float* __restrict__ lsnr_out) {
  __shared__ float sWh[HS_ * 3 * HS_];
  const int tid = threadIdx.x;
  for (int i = tid; i < HS_ * 3 * HS_; i += blockDim.x) sWh[i] = Wh[i];
  __syncthreads();
  if (tid >= B_) return;
  float h[HS_];
#pragma unroll
  for (int c = 0; c < HS_; ++c) h[c] = 0.f;
  for (int t = 0; t < T_; ++t) {
    const float* xr = xw + ((size_t)t * B_ + tid) * 48;
    float hn[HS_];
#pragma unroll
    for (int j = 0; j < HS_; ++j) {
      float b0 = bh[j], b1 = bh[16 + j], b2 = bh[32 + j];
#pragma unroll
      for (int c = 0; c < HS_; ++c) {
        float hc = h[c];
        b0 += hc * sWh[c * 48 + j];
        b1 += hc * sWh[c * 48 + 16 + j];
        b2 += hc * sWh[c * 48 + 32 + j];
      }
      float r = sigm(xr[j] + b0), z = sigm(xr[16 + j] + b1);
      float n = tanhf(xr[32 + j] + r * b2);
      hn[j] = (1.f - z) * n + z * h[j];
    }
#pragma unroll
    for (int c = 0; c < HS_; ++c) h[c] = hn[c];
    float acc = fcb[0];
#pragma unroll
    for (int c = 0; c < HS_; ++c) acc += h[c] * fcW[c];
    float l = sigm(acc) * 50.f - 15.f;
    lsnr_ws[tid * T_ + t]  = l;
    lsnr_out[tid * T_ + t] = l;
  }
}

// ---------------- ERB mask head: m = sigmoid(conv(LN(x))) ------------------
__global__ void k_erb_out(const float* __restrict__ X, const float* __restrict__ g,
                          const float* __restrict__ be, const float* __restrict__ Wco,
                          const float* __restrict__ bco, float* __restrict__ mout) {
  __shared__ float Xs[E_ * C_];
  __shared__ float stats[32];
  const int lane = threadIdx.x;               // lane == e
  const float* xg = X + (size_t)blockIdx.x * E_ * C_;
  for (int i = lane; i < E_ * C_; i += 32) Xs[i] = xg[i];
  __syncthreads();
  if (lane < C_) {
    float s = 0, s2 = 0;
    for (int f = 0; f < E_; ++f) { float v = Xs[f * C_ + lane]; s += v; s2 += v * v; }
    float mu = s / E_;
    stats[lane] = mu;
    stats[16 + lane] = rsqrtf(s2 / E_ - mu * mu + 1e-5f);
  }
  __syncthreads();
  float acc = bco[0];
#pragma unroll
  for (int c = 0; c < C_; ++c) {
    float ln = (Xs[lane * C_ + c] - stats[c]) * stats[16 + c] * g[lane] + be[lane];
    acc += ln * Wco[c];
  }
  mout[(size_t)blockIdx.x * E_ + lane] = sigm(acc);
}

// ---------------- apply erb mask to spec, build spec_f ---------------------
__global__ void k_apply_mask(const float* __restrict__ spec, const float* __restrict__ m,
                             const float* __restrict__ fb, float* __restrict__ spec_out,
                             float* __restrict__ specf) {
  __shared__ float ms[E_];
  const int bt = blockIdx.x, b = bt / T_, t = bt % T_;
  const int tid = threadIdx.x;
  if (tid < E_) ms[tid] = m[(size_t)bt * E_ + tid];
  __syncthreads();
  for (int f = tid; f < FREQ_; f += blockDim.x) {
    float mf = 0.f;
    for (int e = 0; e < E_; ++e) mf += ms[e] * fb[e * FREQ_ + f];
    size_t idx = ((size_t)bt * FREQ_ + f) * 2;
    float re = spec[idx] * mf, im = spec[idx + 1] * mf;
    spec_out[idx] = re;
    spec_out[idx + 1] = im;
    if (f < DF_) {
      specf[(((size_t)b * 2 + 0) * T_ + t) * DF_ + f] = re;
      specf[(((size_t)b * 2 + 1) * T_ + t) * DF_ + f] = im;
    }
  }
}

// ---------------- per-stage refinement mask --------------------------------
__global__ void k_mask(const float* __restrict__ lsnr, float lim, float* __restrict__ mask) {
  int i = blockIdx.x * blockDim.x + threadIdx.x;
  if (i < B_ * T_) {
    float v = lsnr[i];
    mask[i] = (v < lim && v > -10.f) ? 1.f : 0.f;
  }
}

// ---------------- refinement input: LN(DF) -> conv1x1(2->16) -> gelu -------
__global__ void k_ref_in(const float* __restrict__ specf, const float* __restrict__ g,
                         const float* __restrict__ be, const float* __restrict__ Wci,
                         const float* __restrict__ bci, float* __restrict__ X) {
  __shared__ float l0[DF_], l1[DF_];
  const int bt = blockIdx.x, b = bt / T_, t = bt % T_;
  const int lane = threadIdx.x;
  for (int r = 0; r < 2; ++r) {
    const float* sp = specf + (((size_t)b * 2 + r) * T_ + t) * DF_;
    float s = 0, s2 = 0;
    for (int f = lane; f < DF_; f += 32) { float v = sp[f]; s += v; s2 += v * v; }
    for (int o = 16; o > 0; o >>= 1) { s += __shfl_xor(s, o, 32); s2 += __shfl_xor(s2, o, 32); }
    float mu = s / DF_;
    float rs = rsqrtf(s2 / DF_ - mu * mu + 1e-5f);
    float* dst = r ? l1 : l0;
    for (int f = lane; f < DF_; f += 32) dst[f] = (sp[f] - mu) * rs * g[f] + be[f];
  }
  __syncthreads();
  float* xg = X + (size_t)bt * DF_ * C_;
  for (int i = lane; i < DF_ * C_; i += 32) {
    int f = i >> 4, c = i & 15;
    xg[i] = gelu_f(l0[f] * Wci[c] + l1[f] * Wci[C_ + c] + bci[c]);
  }
}

// ---------------- refinement output head + masked residual + outs ----------
__global__ void k_ref_out(const float* __restrict__ X, const float* __restrict__ g,
                          const float* __restrict__ be, const float* __restrict__ Wco,
                          const float* __restrict__ bco, const float* __restrict__ mask,
                          float* __restrict__ specf, float* __restrict__ outs) {
  __shared__ float Xs[DF_ * C_];
  __shared__ float stats[32];
  const int bt = blockIdx.x, b = bt / T_, t = bt % T_;
  const int lane = threadIdx.x;
  const float* xg = X + (size_t)bt * DF_ * C_;
  for (int i = lane; i < DF_ * C_; i += 32) Xs[i] = xg[i];
  __syncthreads();
  if (lane < C_) {
    float s = 0, s2 = 0;
    for (int f = 0; f < DF_; ++f) { float v = Xs[f * C_ + lane]; s += v; s2 += v * v; }
    float mu = s / DF_;
    stats[lane] = mu;
    stats[16 + lane] = rsqrtf(s2 / DF_ - mu * mu + 1e-5f);
  }
  __syncthreads();
  const float mk = mask[bt];
  for (int i = lane; i < DF_ * 2; i += 32) {
    int f = i >> 1, r = i & 1;
    float acc = bco[r];
#pragma unroll
    for (int c = 0; c < C_; ++c) {
      float ln = (Xs[f * C_ + c] - stats[c]) * stats[16 + c] * g[f] + be[f];
      acc += ln * Wco[c * 2 + r];
    }
    float y = tanhf(acc);
    size_t si = (((size_t)b * 2 + r) * T_ + t) * DF_ + f;
    float v = specf[si];
    if (mk > 0.5f) { v += y; specf[si] = v; }
    outs[((size_t)bt * DF_ + f) * 2 + r] = v;    // outs[s]: [B,1,T,DF,2]
  }
}

// ---------------- write refined band back into spec ------------------------
__global__ void k_finalize(const float* __restrict__ specf, float* __restrict__ spec_out) {
  int i = blockIdx.x * blockDim.x + threadIdx.x;
  const int total = B_ * T_ * DF_ * 2;
  if (i >= total) return;
  int r = i & 1;
  int f = (i >> 1) % DF_;
  int bt = (i >> 1) / DF_;
  int b = bt / T_, t = bt % T_;
  spec_out[((size_t)bt * FREQ_ + f) * 2 + r] =
      specf[(((size_t)b * 2 + r) * T_ + t) * DF_ + f];
}

// ============================ host side ====================================
extern "C" void kernel_launch(void* const* d_in, const int* in_sizes, int n_in,
                              void* d_out, int out_size, void* d_ws, size_t ws_size,
                              hipStream_t stream) {
  (void)in_sizes; (void)n_in; (void)out_size; (void)ws_size;
  const float* spec     = (const float*)d_in[0];
  const float* feat_erb = (const float*)d_in[1];
  // d_in[2] = feat_spec: unused by the reference network body
  const float* erb_inv  = (const float*)d_in[3];
  const float* e_ln_in_g = (const float*)d_in[4];
  const float* e_ln_in_b = (const float*)d_in[5];
  const float* e_Wci     = (const float*)d_in[6];
  const float* e_bci     = (const float*)d_in[7];
  const float* e_gru_Wi  = (const float*)d_in[8];
  const float* e_gru_Wh  = (const float*)d_in[9];
  const float* e_ln_a_g  = (const float*)d_in[10];
  const float* e_ln_a_b  = (const float*)d_in[11];
  const float* e_Wq      = (const float*)d_in[12];
  const float* e_Wk      = (const float*)d_in[13];
  const float* e_Wv      = (const float*)d_in[14];
  const float* e_Wo      = (const float*)d_in[15];
  const float* e_Wsh     = (const float*)d_in[16];
  const float* e_ln_m_g  = (const float*)d_in[17];
  const float* e_ln_m_b  = (const float*)d_in[18];
  const float* e_Wm1     = (const float*)d_in[19];
  const float* e_bm1     = (const float*)d_in[20];
  const float* e_Wm2     = (const float*)d_in[21];
  const float* e_bm2     = (const float*)d_in[22];
  const float* e_ln_o_g  = (const float*)d_in[23];
  const float* e_ln_o_b  = (const float*)d_in[24];
  const float* e_Wco     = (const float*)d_in[25];
  const float* e_bco     = (const float*)d_in[26];
  const float* snr_Wi    = (const float*)d_in[27];
  const float* snr_Wh    = (const float*)d_in[28];
  const float* snr_bi    = (const float*)d_in[29];
  const float* snr_bh    = (const float*)d_in[30];
  const float* fc_W      = (const float*)d_in[31];
  const float* fc_b      = (const float*)d_in[32];
  const float* r_ln_in_g = (const float*)d_in[33];
  const float* r_ln_in_b = (const float*)d_in[34];
  const float* r_Wci     = (const float*)d_in[35];
  const float* r_bci     = (const float*)d_in[36];
  const float* r_gru_Wi  = (const float*)d_in[37];
  const float* r_gru_Wh  = (const float*)d_in[38];
  const float* r_ln_a_g  = (const float*)d_in[39];
  const float* r_ln_a_b  = (const float*)d_in[40];
  const float* r_Wq      = (const float*)d_in[41];
  const float* r_Wk      = (const float*)d_in[42];
  const float* r_Wv      = (const float*)d_in[43];
  const float* r_Wo      = (const float*)d_in[44];
  const float* r_Wsh     = (const float*)d_in[45];
  const float* r_ln_m_g  = (const float*)d_in[46];
  const float* r_ln_m_b  = (const float*)d_in[47];
  const float* r_Wm1     = (const float*)d_in[48];
  const float* r_bm1     = (const float*)d_in[49];
  const float* r_Wm2     = (const float*)d_in[50];
  const float* r_bm2     = (const float*)d_in[51];
  const float* r_ln_o_g  = (const float*)d_in[52];
  const float* r_ln_o_b  = (const float*)d_in[53];
  const float* r_Wco     = (const float*)d_in[54];
  const float* r_bco     = (const float*)d_in[55];

  // ---- workspace carve (fp32 region then packed f16 weights), ~57 MB ----
  float* ws      = (float*)d_ws;
  float* Xws     = ws;                                       // B*T*96*16
  float* specf   = Xws + (size_t)B_ * T_ * DF_ * C_;         // B*2*T*96
  float* xw_snr  = specf + (size_t)B_ * 2 * T_ * DF_;        // B*T*48
  float* lsnr_ws = xw_snr + (size_t)B_ * T_ * 48;            // B*T
  float* mask_ws = lsnr_ws + (size_t)B_ * T_;                // B*T
  half_t* hpack  = (half_t*)(mask_ws + (size_t)B_ * T_);     // packed weight frags
  // per-stage block: Wq(2048) Wk(2048) Wv(2048) Wo(1024) Wm1(512) Wm2(512) = 8192 halves
  auto WqF  = [&](int s) { return hpack + (size_t)s * 8192; };
  auto WkF  = [&](int s) { return hpack + (size_t)s * 8192 + 2048; };
  auto WvF  = [&](int s) { return hpack + (size_t)s * 8192 + 4096; };
  auto WoF  = [&](int s) { return hpack + (size_t)s * 8192 + 6144; };
  auto Wm1F = [&](int s) { return hpack + (size_t)s * 8192 + 7168; };
  auto Wm2F = [&](int s) { return hpack + (size_t)s * 8192 + 7680; };
  half_t* snrWiF = hpack + (size_t)5 * 8192;                 // 24576 halves

  // output layout: spec | m | lsnr | outs (flat, return order)
  float* out_spec = (float*)d_out;
  float* out_m    = out_spec + (size_t)B_ * T_ * FREQ_ * 2;
  float* out_lsnr = out_m + (size_t)B_ * T_ * E_;
  float* out_outs = out_lsnr + (size_t)B_ * T_;

  const int BT = B_ * T_;
  auto smA = [](int F) -> size_t {
    return (size_t)4 * (32 * F + 32) + (size_t)2 * 512 * (F >> 4) * (9 + (F >> 5));
  };
  auto smM = [](int F) -> size_t {
    return (size_t)4 * (16 * F + 32) + (size_t)2 * 2 * 512 * (F >> 4);
  };
  hipFuncSetAttribute((const void*)k_attn,
                      hipFuncAttributeMaxDynamicSharedMemorySize, (int)smA(DF_) + 256);

  // ---- pack all GEMM weights into f16 B-fragment layout (once per call) ----
  for (int s = 0; s < 5; ++s) {
    const float* wq = s == 0 ? e_Wq : r_Wq + (s - 1) * C_ * DH_;
    const float* wk = s == 0 ? e_Wk : r_Wk + (s - 1) * C_ * DH_;
    const float* wv = s == 0 ? e_Wv : r_Wv + (s - 1) * C_ * DH_;
    const float* wo = s == 0 ? e_Wo : r_Wo + (s - 1) * DH_ * C_;
    const float* w1 = s == 0 ? e_Wm1 : r_Wm1 + (s - 1) * C_ * C_;
    const float* w2 = s == 0 ? e_Wm2 : r_Wm2 + (s - 1) * C_ * C_;
    k_pack_b<<<4, 512, 0, stream>>>(wq, WqF(s), C_, DH_);
    k_pack_b<<<4, 512, 0, stream>>>(wk, WkF(s), C_, DH_);
    k_pack_b<<<4, 512, 0, stream>>>(wv, WvF(s), C_, DH_);
    k_pack_b<<<2, 512, 0, stream>>>(wo, WoF(s), DH_, C_);
    k_pack_b<<<1, 512, 0, stream>>>(w1, Wm1F(s), C_, C_);
    k_pack_b<<<1, 512, 0, stream>>>(w2, Wm2F(s), C_, C_);
  }
  k_pack_b<<<48, 512, 0, stream>>>(snr_Wi, snrWiF, C_ * E_, 48);

  // -------- ERB stage --------
  k_erb_in<<<BT, 32, 0, stream>>>(feat_erb, e_ln_in_g, e_ln_in_b, e_Wci, e_bci, Xws);
  k_gru<<<1, 256, 0, stream>>>(Xws, e_gru_Wi, e_gru_Wh, nullptr, E_);   // B*E = 256 rows
  k_attn<<<BT, 32, smA(E_), stream>>>(Xws, E_, e_ln_a_g, e_ln_a_b,
                                      WqF(0), WkF(0), WvF(0), WoF(0), e_Wsh);
  k_mlp<<<BT, 32, smM(E_), stream>>>(Xws, E_, e_ln_m_g, e_ln_m_b,
                                     Wm1F(0), e_bm1, Wm2F(0), e_bm2);
  k_snr_xw<<<dim3(BT / 16, 3), 32, 0, stream>>>(Xws, snrWiF, snr_bi, xw_snr);
  k_snr_gru<<<1, 32, 0, stream>>>(xw_snr, snr_Wh, snr_bh, fc_W, fc_b, lsnr_ws, out_lsnr);
  k_erb_out<<<BT, 32, 0, stream>>>(Xws, e_ln_o_g, e_ln_o_b, e_Wco, e_bco, out_m);
  k_apply_mask<<<BT, 256, 0, stream>>>(spec, out_m, erb_inv, out_spec, specf);

  // -------- refinement stages --------
  const float lims[4] = {20.f, 10.f, 5.f, 0.f};
  for (int s = 0; s < 4; ++s) {
    k_mask<<<(BT + 255) / 256, 256, 0, stream>>>(lsnr_ws, lims[s], mask_ws);
    k_ref_in<<<BT, 32, 0, stream>>>(specf, r_ln_in_g + s * DF_, r_ln_in_b + s * DF_,
                                    r_Wci + s * 2 * C_, r_bci + s * C_, Xws);
    k_gru<<<(B_ * DF_) / 256, 256, 0, stream>>>(Xws, r_gru_Wi + s * C_ * 3 * C_,
                                                r_gru_Wh + s * C_ * 3 * C_, mask_ws, DF_);
    k_attn<<<BT, 32, smA(DF_), stream>>>(Xws, DF_, r_ln_a_g + s * DF_, r_ln_a_b + s * DF_,
                                         WqF(s + 1), WkF(s + 1), WvF(s + 1), WoF(s + 1),
                                         r_Wsh + s * DH_ * 9);
    k_mlp<<<BT, 32, smM(DF_), stream>>>(Xws, DF_, r_ln_m_g + s * DF_, r_ln_m_b + s * DF_,
                                        Wm1F(s + 1), r_bm1 + s * C_,
                                        Wm2F(s + 1), r_bm2 + s * C_);
    k_ref_out<<<BT, 32, 0, stream>>>(Xws, r_ln_o_g + s * DF_, r_ln_o_b + s * DF_,
                                     r_Wco + s * C_ * 2, r_bco + s * 2, mask_ws, specf,
                                     out_outs + (size_t)s * BT * DF_ * 2);
  }
  k_finalize<<<((B_ * T_ * DF_ * 2) + 255) / 256, 256, 0, stream>>>(specf, out_spec);
}
